// MultiHeadAttention_50079318672059
// MI455X (gfx1250) — matrix operand order repaired and checked by reading, not compiled, not run on hardware
//
#include <hip/hip_runtime.h>

typedef __bf16 bf16_t;
typedef __attribute__((ext_vector_type(16))) __bf16 v16bf;
typedef __attribute__((ext_vector_type(8)))  float v8f;

#define BATCH 4
#define SEQ   2048
#define DIM   768
#define NH    12
#define DHEAD 64
#define QKVN  2304
#define SCALE 0.125f

union AFrag { v16bf v; unsigned int u[8]; };
union CFrag { v8f v; float f[8]; };

static __device__ __forceinline__ v8f wmma_bf16(v16bf a, v16bf b, v8f c) {
  return __builtin_amdgcn_wmma_f32_16x16x32_bf16(false, a, false, b, (short)0, c, false, false);
}
static __device__ __forceinline__ bf16_t f2bf(float f) { return (bf16_t)f; }

// ---------------------------------------------------------------------------
// Kernel 1: QKV projection GEMM  C[8192,2304] = X[8192,768] * Wqkv[768,2304]
// Double-buffered LDS (one barrier per k-step), global loads for step k+1
// issued before the WMMA block for step k. Output scattered as bf16 into
// per-head [b,h,n,64] q/k/v buffers. 256 threads (8 waves), tile 128x128.
// ---------------------------------------------------------------------------
__global__ __launch_bounds__(256) void qkv_gemm_kernel(
    const float* __restrict__ x, const float* __restrict__ wqkv,
    bf16_t* __restrict__ qh, bf16_t* __restrict__ kh, bf16_t* __restrict__ vh)
{
  __shared__ alignas(16) bf16_t As[2][128][32];   // [buf][row][k]
  __shared__ alignas(16) bf16_t Bs[2][128][32];   // [buf][col][k] (transposed)
  const int tid = threadIdx.x;
  const int lane = tid & 31, wave = tid >> 5;
  const int half = lane >> 4, lr = lane & 15;
  const int wr = wave >> 2, wc = wave & 3;
  const int tm = blockIdx.x, tn = blockIdx.y;

  float4 ar[4], br[4];
  auto load_tiles = [&](int k0) {
    #pragma unroll
    for (int p = 0; p < 4; ++p)
      ar[p] = *(const float4*)(x + (size_t)(tm * 128 + (tid >> 3) + p * 32) * DIM
                                 + k0 + (tid & 7) * 4);
    #pragma unroll
    for (int p = 0; p < 4; ++p)
      br[p] = *(const float4*)(wqkv + (size_t)(k0 + (tid >> 5) + p * 8) * QKVN
                                    + tn * 128 + (tid & 31) * 4);
  };
  auto store_tiles = [&](int b) {
    #pragma unroll
    for (int p = 0; p < 4; ++p) {
      const int r = (tid >> 3) + p * 32, c4 = (tid & 7) * 4;
      As[b][r][c4 + 0] = f2bf(ar[p].x); As[b][r][c4 + 1] = f2bf(ar[p].y);
      As[b][r][c4 + 2] = f2bf(ar[p].z); As[b][r][c4 + 3] = f2bf(ar[p].w);
    }
    #pragma unroll
    for (int p = 0; p < 4; ++p) {
      const int kr = (tid >> 5) + p * 8, c4 = (tid & 31) * 4;
      Bs[b][c4 + 0][kr] = f2bf(br[p].x); Bs[b][c4 + 1][kr] = f2bf(br[p].y);
      Bs[b][c4 + 2][kr] = f2bf(br[p].z); Bs[b][c4 + 3][kr] = f2bf(br[p].w);
    }
  };

  CFrag acc[4][2];
  for (int a = 0; a < 4; ++a)
    for (int b = 0; b < 2; ++b)
      for (int i = 0; i < 8; ++i) acc[a][b].f[i] = 0.f;

  load_tiles(0);
  store_tiles(0);
  __syncthreads();

  for (int ks = 0; ks < 24; ++ks) {
    const int cur = ks & 1;
    if (ks + 1 < 24) load_tiles((ks + 1) * 32);

    AFrag bfr[2];
    #pragma unroll
    for (int ct = 0; ct < 2; ++ct) {
      const int col = wc * 32 + ct * 16 + lr;
      const int kb2 = half * 16;
      #pragma unroll
      for (int j = 0; j < 8; ++j)
        bfr[ct].u[j] = *(const unsigned int*)&Bs[cur][col][kb2 + 2 * j];
    }
    #pragma unroll
    for (int rt = 0; rt < 4; ++rt) {
      AFrag a;
      const int row = wr * 64 + rt * 16 + lr;
      const int kb = half * 16;
      #pragma unroll
      for (int j = 0; j < 4; ++j) {
        a.u[j]     = *(const unsigned int*)&As[cur][row][kb + 2 * j];
        a.u[4 + j] = *(const unsigned int*)&As[cur][row][kb + 8 + 2 * j];
      }
      acc[rt][0].v = wmma_bf16(a.v, bfr[0].v, acc[rt][0].v);
      acc[rt][1].v = wmma_bf16(a.v, bfr[1].v, acc[rt][1].v);
    }

    if (ks + 1 < 24) store_tiles(1 - cur);
    __syncthreads();
  }

  // epilogue: scatter bf16 to q/k/v [b,h,n,d] (no integer division)
  for (int rt = 0; rt < 4; ++rt)
    for (int ct = 0; ct < 2; ++ct) {
      #pragma unroll
      for (int i = 0; i < 8; ++i) {
        const int row  = tm * 128 + wr * 64 + rt * 16 + i + 8 * half;
        const int coln = tn * 128 + wc * 32 + ct * 16 + lr;
        const int which = (coln >= 1536) ? 2 : ((coln >= 768) ? 1 : 0);
        const int rem = coln - which * DIM;
        const int h = rem >> 6, d = rem & 63;
        const int bi = row >> 11, nr = row & 2047;
        const size_t dst = ((size_t)(bi * NH + h) * SEQ + nr) * DHEAD + d;
        const bf16_t val = f2bf(acc[rt][ct].f[i]);
        if (which == 0)      qh[dst] = val;
        else if (which == 1) kh[dst] = val;
        else                 vh[dst] = val;
      }
    }
}

// ---------------------------------------------------------------------------
// Kernel 2: flash attention. 128 threads (4 waves), 64 q-rows per block for
// one (b,h); each wave owns 16 q-rows. Online softmax over k-tiles of 64
// keys; scores never leave registers. V tile double-buffered in LDS (one
// barrier per k-tile); K rows + mask rows for the next tile prefetched.
// ---------------------------------------------------------------------------
__global__ __launch_bounds__(128) void flash_attn_kernel(
    const bf16_t* __restrict__ qh, const bf16_t* __restrict__ kh,
    const bf16_t* __restrict__ vh, const int* __restrict__ mask,
    bf16_t* __restrict__ ctx)
{
  __shared__ alignas(16) bf16_t Vt[2][64][64];   // [buf][dim][key]
  __shared__ alignas(16) bf16_t Pl[4][16][64];   // per-wave P staging [row][key]
  const int tid = threadIdx.x;
  const int lane = tid & 31, wave = tid >> 5;
  const int half = lane >> 4, lr = lane & 15;
  const int bh = blockIdx.y;
  const int bi = bh / NH, h = bh % NH;
  const int qrow0 = blockIdx.x * 64 + wave * 16;

  const int vkey = tid & 63;
  const int vdh0 = (tid >> 6) * 32;
  unsigned int vr[16];
  auto load_v = [&](int kb) {
    const bf16_t* vp = vh + ((size_t)bh * SEQ + kb + vkey) * DHEAD + vdh0;
    #pragma unroll
    for (int j = 0; j < 16; ++j)
      vr[j] = *(const unsigned int*)(vp + 2 * j);
  };
  auto store_v = [&](int b) {
    union { unsigned int u; bf16_t h2[2]; } t2;
    #pragma unroll
    for (int j = 0; j < 16; ++j) {
      t2.u = vr[j];
      Vt[b][vdh0 + 2 * j][vkey]     = t2.h2[0];
      Vt[b][vdh0 + 2 * j + 1][vkey] = t2.h2[1];
    }
  };

  // Q fragments (A-layout: lane holds row lr, k = 16*half + s, contiguous)
  AFrag qa[2];
  {
    const bf16_t* qp = qh + ((size_t)bh * SEQ + qrow0 + lr) * DHEAD;
    const int kb = half * 16;
    #pragma unroll
    for (int f = 0; f < 2; ++f)
      #pragma unroll
      for (int j = 0; j < 4; ++j) {
        qa[f].u[j]     = *(const unsigned int*)(qp + f * 32 + kb + 2 * j);
        qa[f].u[4 + j] = *(const unsigned int*)(qp + f * 32 + kb + 8 + 2 * j);
      }
  }

  float mrow[8], lrow[8];
  CFrag O[4];
  #pragma unroll
  for (int i = 0; i < 8; ++i) { mrow[i] = -3.0e38f; lrow[i] = 0.f; }
  for (int d = 0; d < 4; ++d)
    for (int i = 0; i < 8; ++i) O[d].f[i] = 0.f;

  load_v(0);
  store_v(0);
  __syncthreads();

  for (int kt = 0; kt < 32; ++kt) {
    const int kb0 = kt * 64;
    const int cur = kt & 1;

    if (kt + 1 < 32) {
      const int kb1 = kb0 + 64;
      load_v(kb1);
      // prefetch next K tile rows and next mask segment (L2-resident streams)
      #pragma unroll
      for (int sub = 0; sub < 4; ++sub)
        __builtin_prefetch(kh + ((size_t)bh * SEQ + kb1 + sub * 16 + lr) * DHEAD
                              + half * 32, 0, 1);
      #pragma unroll
      for (int i = 0; i < 8; ++i)
        __builtin_prefetch(mask + ((size_t)bi * SEQ + qrow0 + i + 8 * half) * SEQ
                               + kb1 + lr * 4, 0, 1);
    }

    // S = Q * K^T  (B-frag: lane = key column, K = head-dim, contiguous)
    CFrag S[4];
    for (int s = 0; s < 4; ++s)
      for (int i = 0; i < 8; ++i) S[s].f[i] = 0.f;
    #pragma unroll
    for (int sub = 0; sub < 4; ++sub) {
      #pragma unroll
      for (int f = 0; f < 2; ++f) {
        AFrag kf;
        const bf16_t* kp = kh + ((size_t)bh * SEQ + kb0 + sub * 16 + lr) * DHEAD
                              + f * 32 + half * 16;
        #pragma unroll
        for (int j = 0; j < 8; ++j)
          kf.u[j] = *(const unsigned int*)(kp + 2 * j);
        S[sub].v = wmma_bf16(qa[f].v, kf.v, S[sub].v);
      }
    }

    // mask + online softmax (row i+8*half lives in one VGPR slot across a
    // 16-lane half -> reduce with shfl_xor 1,2,4,8)
    #pragma unroll
    for (int i = 0; i < 8; ++i) {
      const int qrow = qrow0 + i + 8 * half;
      const int* mp = mask + ((size_t)bi * SEQ + qrow) * SEQ + kb0;
      #pragma unroll
      for (int sub = 0; sub < 4; ++sub) {
        float s = S[sub].f[i] * SCALE;
        if (mp[sub * 16 + lr] == 0) s = -1.0e9f;
        S[sub].f[i] = s;
      }
      float mx = fmaxf(fmaxf(S[0].f[i], S[1].f[i]), fmaxf(S[2].f[i], S[3].f[i]));
      for (int o = 1; o < 16; o <<= 1) mx = fmaxf(mx, __shfl_xor(mx, o, 32));
      const float mnew = fmaxf(mrow[i], mx);
      const float alpha = __expf(mrow[i] - mnew);
      float rs = 0.f;
      #pragma unroll
      for (int sub = 0; sub < 4; ++sub) {
        const float p = __expf(S[sub].f[i] - mnew);
        S[sub].f[i] = p;
        rs += p;
      }
      for (int o = 1; o < 16; o <<= 1) rs += __shfl_xor(rs, o, 32);
      lrow[i] = lrow[i] * alpha + rs;
      mrow[i] = mnew;
      #pragma unroll
      for (int d = 0; d < 4; ++d) O[d].f[i] *= alpha;
    }

    // redistribute P: C-layout -> LDS -> A-layout (per-wave region; DScnt
    // ordering is within-wave, compiler inserts s_wait_dscnt)
    #pragma unroll
    for (int i = 0; i < 8; ++i)
      #pragma unroll
      for (int sub = 0; sub < 4; ++sub)
        Pl[wave][i + 8 * half][sub * 16 + lr] = f2bf(S[sub].f[i]);

    AFrag pf[2];
    {
      const int kb = half * 16;
      #pragma unroll
      for (int f = 0; f < 2; ++f)
        #pragma unroll
        for (int j = 0; j < 4; ++j) {
          pf[f].u[j]     = *(const unsigned int*)&Pl[wave][lr][f * 32 + kb + 2 * j];
          pf[f].u[4 + j] = *(const unsigned int*)&Pl[wave][lr][f * 32 + kb + 8 + 2 * j];
        }
    }

    // O += P * V   (B-frag: lane = dim column, K = key, contiguous in Vt row)
    #pragma unroll
    for (int dsub = 0; dsub < 4; ++dsub) {
      #pragma unroll
      for (int f = 0; f < 2; ++f) {
        AFrag bv;
        const int keyb = f * 32 + half * 16;
        #pragma unroll
        for (int j = 0; j < 8; ++j)
          bv.u[j] = *(const unsigned int*)&Vt[cur][dsub * 16 + lr][keyb + 2 * j];
        O[dsub].v = wmma_bf16(pf[f].v, bv.v, O[dsub].v);
      }
    }

    if (kt + 1 < 32) store_v(1 - cur);
    __syncthreads();
  }

  // normalize and write context as bf16, layout [b, n, h*64]
  #pragma unroll
  for (int dsub = 0; dsub < 4; ++dsub)
    #pragma unroll
    for (int i = 0; i < 8; ++i) {
      const int qrow = qrow0 + i + 8 * half;
      const float ov = O[dsub].f[i] / lrow[i];
      ctx[((size_t)(bi * SEQ + qrow)) * DIM + h * DHEAD + dsub * 16 + lr] = f2bf(ov);
    }
}

// ---------------------------------------------------------------------------
// Kernel 3: output projection  out[8192,768] = ctx[8192,768] * Wout[768,768]
// Same double-buffered structure as kernel 1; A is already bf16.
// ---------------------------------------------------------------------------
__global__ __launch_bounds__(256) void out_gemm_kernel(
    const bf16_t* __restrict__ ctx, const float* __restrict__ wout,
    float* __restrict__ out)
{
  __shared__ alignas(16) bf16_t As[2][128][32];
  __shared__ alignas(16) bf16_t Bs[2][128][32];
  const int tid = threadIdx.x;
  const int lane = tid & 31, wave = tid >> 5;
  const int half = lane >> 4, lr = lane & 15;
  const int wr = wave >> 2, wc = wave & 3;
  const int tm = blockIdx.x, tn = blockIdx.y;

  unsigned int au[8];
  float4 br[4];
  const int arow = tid >> 1;
  const int auo = (tid & 1) * 8;
  auto load_tiles = [&](int k0) {
    const unsigned int* ap =
        (const unsigned int*)(ctx + (size_t)(tm * 128 + arow) * DIM + k0) + auo;
    #pragma unroll
    for (int j = 0; j < 8; ++j) au[j] = ap[j];
    #pragma unroll
    for (int p = 0; p < 4; ++p)
      br[p] = *(const float4*)(wout + (size_t)(k0 + (tid >> 5) + p * 8) * DIM
                                    + tn * 128 + (tid & 31) * 4);
  };
  auto store_tiles = [&](int b) {
    unsigned int* dp = (unsigned int*)&As[b][arow][0] + auo;
    #pragma unroll
    for (int j = 0; j < 8; ++j) dp[j] = au[j];
    #pragma unroll
    for (int p = 0; p < 4; ++p) {
      const int kr = (tid >> 5) + p * 8, c4 = (tid & 31) * 4;
      Bs[b][c4 + 0][kr] = f2bf(br[p].x); Bs[b][c4 + 1][kr] = f2bf(br[p].y);
      Bs[b][c4 + 2][kr] = f2bf(br[p].z); Bs[b][c4 + 3][kr] = f2bf(br[p].w);
    }
  };

  CFrag acc[4][2];
  for (int a = 0; a < 4; ++a)
    for (int b = 0; b < 2; ++b)
      for (int i = 0; i < 8; ++i) acc[a][b].f[i] = 0.f;

  load_tiles(0);
  store_tiles(0);
  __syncthreads();

  for (int ks = 0; ks < 24; ++ks) {
    const int cur = ks & 1;
    if (ks + 1 < 24) load_tiles((ks + 1) * 32);

    AFrag bfr[2];
    #pragma unroll
    for (int ct = 0; ct < 2; ++ct) {
      const int col = wc * 32 + ct * 16 + lr;
      const int kb2 = half * 16;
      #pragma unroll
      for (int j = 0; j < 8; ++j)
        bfr[ct].u[j] = *(const unsigned int*)&Bs[cur][col][kb2 + 2 * j];
    }
    #pragma unroll
    for (int rt = 0; rt < 4; ++rt) {
      AFrag a;
      const int row = wr * 64 + rt * 16 + lr;
      const int kb = half * 16;
      #pragma unroll
      for (int j = 0; j < 4; ++j) {
        a.u[j]     = *(const unsigned int*)&As[cur][row][kb + 2 * j];
        a.u[4 + j] = *(const unsigned int*)&As[cur][row][kb + 8 + 2 * j];
      }
      acc[rt][0].v = wmma_bf16(a.v, bfr[0].v, acc[rt][0].v);
      acc[rt][1].v = wmma_bf16(a.v, bfr[1].v, acc[rt][1].v);
    }

    if (ks + 1 < 24) store_tiles(1 - cur);
    __syncthreads();
  }

  for (int rt = 0; rt < 4; ++rt)
    for (int ct = 0; ct < 2; ++ct) {
      #pragma unroll
      for (int i = 0; i < 8; ++i) {
        const int row  = tm * 128 + wr * 64 + rt * 16 + i + 8 * half;
        const int coln = tn * 128 + wc * 32 + ct * 16 + lr;
        out[(size_t)row * DIM + coln] = acc[rt][ct].f[i];
      }
    }
}

extern "C" void kernel_launch(void* const* d_in, const int* in_sizes, int n_in,
                              void* d_out, int out_size, void* d_ws, size_t ws_size,
                              hipStream_t stream) {
  const float* x    = (const float*)d_in[0];
  const int*   mask = (const int*)d_in[1];
  const float* wqkv = (const float*)d_in[2];
  const float* wout = (const float*)d_in[3];
  float* out = (float*)d_out;

  const size_t per = (size_t)BATCH * NH * SEQ * DHEAD;  // 6,291,456 elements
  bf16_t* qh  = (bf16_t*)d_ws;
  bf16_t* kh  = qh + per;
  bf16_t* vh  = kh + per;
  bf16_t* ctx = vh + per;   // total ~50.3 MB of workspace

  qkv_gemm_kernel<<<dim3(64, 18), 256, 0, stream>>>(x, wqkv, qh, kh, vh);
  flash_attn_kernel<<<dim3(32, 48), 128, 0, stream>>>(qh, kh, vh, mask, ctx);
  out_gemm_kernel<<<dim3(64, 6), 256, 0, stream>>>(ctx, wout, out);

  (void)in_sizes; (void)n_in; (void)out_size; (void)ws_size;
}
